// Model_52518860096402
// MI455X (gfx1250) — compile-verified, matrix operand order checked by reading, not run on hardware
//
#include <hip/hip_runtime.h>

typedef __attribute__((ext_vector_type(16))) __bf16   bf16x16;
typedef __attribute__((ext_vector_type(8)))  float    f32x8;
typedef __attribute__((ext_vector_type(4)))  float    f32x4;
typedef __attribute__((ext_vector_type(4)))  unsigned u32x4;

#define SEQ    2048
#define DHEAD  128
#define KTILE  32      // K/V columns per iteration
#define NWAVE  8
#define QBLK   128     // 16 q-rows per wave * 8 waves

#define KSTR   136     // bf16 elems per K-row in LDS (128 + pad), 272B row, 16B-aligned
#define VSTR   40      // bf16 elems per Vt-row (32 + pad), 80B row, 16B-aligned
#define PSTR   40      // bf16 elems per P-row, 80B row, 16B-aligned

#define LOG2E  1.4426950408889634f

#if defined(__has_builtin)
#  if __has_builtin(__builtin_amdgcn_cvt_pk_bf16_f32)
#    define HAVE_CVT_PK_BF16 1
#  endif
#endif

union FragB { bf16x16 v; unsigned u[8]; u32x4 q[2]; };

__device__ __forceinline__ unsigned pk_bf16x2(float a, float b) {
#ifdef HAVE_CVT_PK_BF16
  auto r = __builtin_amdgcn_cvt_pk_bf16_f32(a, b);
  unsigned u; __builtin_memcpy(&u, &r, sizeof(u));
  return u;
#else
  union { __bf16 h[2]; unsigned u; } x;
  x.h[0] = (__bf16)a;   // fptrunc f32->bf16 (RNE); native cvt on gfx1250
  x.h[1] = (__bf16)b;
  return x.u;
#endif
}

// ---- DPP16 row reductions (16-lane groups), pure VALU: no LDS-unit contention ----
template <int CTRL>
__device__ __forceinline__ float dpp_mov(float x) {
  int i = __float_as_int(x);
  return __int_as_float(__builtin_amdgcn_update_dpp(i, i, CTRL, 0xF, 0xF, false));
}
__device__ __forceinline__ float rowmax16(float x) {
  x = fmaxf(x, dpp_mov<0x128>(x));   // row_ror:8
  x = fmaxf(x, dpp_mov<0x124>(x));   // row_ror:4
  x = fmaxf(x, dpp_mov<0x122>(x));   // row_ror:2
  x = fmaxf(x, dpp_mov<0x121>(x));   // row_ror:1
  return x;
}
__device__ __forceinline__ float rowsum16(float x) {
  x += dpp_mov<0x128>(x);
  x += dpp_mov<0x124>(x);
  x += dpp_mov<0x122>(x);
  x += dpp_mov<0x121>(x);
  return x;
}

// register-staged K/V tile slice (per thread): K already bf16-packed, V raw f32
struct Stage {
  u32x4 kw0, kw1;
  f32x4 v0, v1, v2, v3;
};

__device__ __forceinline__ void stage_load(const float* __restrict__ krp,
                                           const float* __restrict__ vrp,
                                           Stage& s) {
  f32x4 a0 = *(const f32x4*)(krp);
  f32x4 a1 = *(const f32x4*)(krp + 4);
  f32x4 a2 = *(const f32x4*)(krp + 8);
  f32x4 a3 = *(const f32x4*)(krp + 12);
  s.kw0 = (u32x4){ pk_bf16x2(a0.x, a0.y), pk_bf16x2(a0.z, a0.w),
                   pk_bf16x2(a1.x, a1.y), pk_bf16x2(a1.z, a1.w) };
  s.kw1 = (u32x4){ pk_bf16x2(a2.x, a2.y), pk_bf16x2(a2.z, a2.w),
                   pk_bf16x2(a3.x, a3.y), pk_bf16x2(a3.z, a3.w) };
  s.v0 = *(const f32x4*)(vrp);
  s.v1 = *(const f32x4*)(vrp + 4);
  s.v2 = *(const f32x4*)(vrp + 8);
  s.v3 = *(const f32x4*)(vrp + 12);
}

__device__ __forceinline__ void stage_store(unsigned short* __restrict__ sKb,
                                            unsigned short* __restrict__ sVb,
                                            int lrow, int ld0, const Stage& s) {
  *(u32x4*)&sKb[lrow * KSTR + ld0]     = s.kw0;
  *(u32x4*)&sKb[lrow * KSTR + ld0 + 8] = s.kw1;
  float vv[16] = { s.v0.x, s.v0.y, s.v0.z, s.v0.w, s.v1.x, s.v1.y, s.v1.z, s.v1.w,
                   s.v2.x, s.v2.y, s.v2.z, s.v2.w, s.v3.x, s.v3.y, s.v3.z, s.v3.w };
#pragma unroll
  for (int i = 0; i < 8; ++i) {
    unsigned u = pk_bf16x2(vv[2 * i], vv[2 * i + 1]);
    sVb[(ld0 + 2 * i) * VSTR + lrow]     = (unsigned short)(u & 0xFFFFu);
    sVb[(ld0 + 2 * i + 1) * VSTR + lrow] = (unsigned short)(u >> 16);
  }
}

__global__ __launch_bounds__(256) void fa_fwd_kernel(
    const float* __restrict__ q, const float* __restrict__ k,
    const float* __restrict__ v, const float* __restrict__ mask,
    float* __restrict__ out)
{
  __shared__ alignas(16) unsigned short sK [2][KTILE * KSTR];     // double-buffered K tile
  __shared__ alignas(16) unsigned short sVt[2][DHEAD * VSTR];     // double-buffered V^T tile
  __shared__ alignas(16) unsigned short sP [NWAVE * 16 * PSTR];   // per-wave prob scratch

  const int tid  = threadIdx.x;
  const int wave = tid >> 5;
  const int lane = tid & 31;
  const int m    = lane & 15;   // row (A-frag) / column (B/C-frag)
  const int h    = lane >> 4;   // half-wave selector (K-dim split per ISA layout)

  const int bh = blockIdx.y;
  const int qw = blockIdx.x * QBLK + wave * 16;

  const float scale = 0.08838834764831845f;   // 1/sqrt(128)

  // ---- preload Q as 4 A-fragments (16x32 bf16 each) ----
  const float* qbase = q + ((size_t)bh * SEQ + qw + m) * DHEAD;
  FragB qf[4];
#pragma unroll
  for (int c = 0; c < 4; ++c) {
    const float* p0 = qbase + c * 32 + h * 8;
    const float* p1 = qbase + c * 32 + 16 + h * 8;
    f32x4 a0 = *(const f32x4*)p0;
    f32x4 a1 = *(const f32x4*)(p0 + 4);
    f32x4 b0 = *(const f32x4*)p1;
    f32x4 b1 = *(const f32x4*)(p1 + 4);
    qf[c].u[0] = pk_bf16x2(a0.x, a0.y); qf[c].u[1] = pk_bf16x2(a0.z, a0.w);
    qf[c].u[2] = pk_bf16x2(a1.x, a1.y); qf[c].u[3] = pk_bf16x2(a1.z, a1.w);
    qf[c].u[4] = pk_bf16x2(b0.x, b0.y); qf[c].u[5] = pk_bf16x2(b0.z, b0.w);
    qf[c].u[6] = pk_bf16x2(b1.x, b1.y); qf[c].u[7] = pk_bf16x2(b1.z, b1.w);
  }

  // ---- running state: lrun is a PER-LANE PARTIAL sum (reduced once in epilogue;
  //      valid because the rescale factor alpha is lane-uniform each iteration) ----
  f32x8 accO[8];
  float mrun[8], lrun[8];
#pragma unroll
  for (int nc = 0; nc < 8; ++nc) {
    f32x8 z = {0.f, 0.f, 0.f, 0.f, 0.f, 0.f, 0.f, 0.f};
    accO[nc] = z;
  }
#pragma unroll
  for (int r = 0; r < 8; ++r) { mrun[r] = -3.0e38f; lrun[r] = 0.f; }

  const float* kbh = k + (size_t)bh * SEQ * DHEAD;
  const float* vbh = v + (size_t)bh * SEQ * DHEAD;
  const int mbase = (qw + 8 * h) * SEQ + m;     // mask[q_row][k_col] base

  // cooperative tile mapping: 256 threads cover 32 rows x 128 dims
  const int lrow = tid >> 3;          // 0..31
  const int ld0  = (tid & 7) * 16;    // 0..112

  unsigned short* pw = &sP[wave * 16 * PSTR];

  // ---- software pipeline: prefetch tile 0 into registers ----
  Stage st;
  stage_load(kbh + (size_t)lrow * DHEAD + ld0,
             vbh + (size_t)lrow * DHEAD + ld0, st);

  int p = 0;
#pragma unroll 2
  for (int t = 0; t < SEQ; t += KTILE) {
    // commit staged tile to LDS buffer p, then immediately issue next tile's
    // global loads so they are in flight during this iteration's compute
    stage_store(sK[p], sVt[p], lrow, ld0, st);
    if (t + KTILE < SEQ) {
      stage_load(kbh + (size_t)(t + KTILE + lrow) * DHEAD + ld0,
                 vbh + (size_t)(t + KTILE + lrow) * DHEAD + ld0, st);
    }
    __syncthreads();   // buffer p fully written (also fences prior reads of p)

    // ---- S = Q * K^T (16 x 32 tile) ----
    f32x8 acc0 = {0.f, 0.f, 0.f, 0.f, 0.f, 0.f, 0.f, 0.f};
    f32x8 acc1 = {0.f, 0.f, 0.f, 0.f, 0.f, 0.f, 0.f, 0.f};
#pragma unroll
    for (int c = 0; c < 4; ++c) {
      FragB kf0, kf1;
      const unsigned short* r0 = &sK[p][m * KSTR + c * 32 + h * 8];
      kf0.q[0] = *(const u32x4*)r0;
      kf0.q[1] = *(const u32x4*)(r0 + 16);
      const unsigned short* r1 = &sK[p][(m + 16) * KSTR + c * 32 + h * 8];
      kf1.q[0] = *(const u32x4*)r1;
      kf1.q[1] = *(const u32x4*)(r1 + 16);
      acc0 = __builtin_amdgcn_wmma_f32_16x16x32_bf16(false, qf[c].v, false, kf0.v,
                                                     (short)0, acc0, false, false);
      acc1 = __builtin_amdgcn_wmma_f32_16x16x32_bf16(false, qf[c].v, false, kf1.v,
                                                     (short)0, acc1, false, false);
    }

    // ---- mask + online softmax (row max broadcast on DPP16; lrun kept partial) ----
    f32x8 av;
#pragma unroll
    for (int r = 0; r < 8; ++r) {
      float s0 = acc0[r] * scale + mask[mbase + r * SEQ + t];
      float s1 = acc1[r] * scale + mask[mbase + r * SEQ + t + 16];
      float mx = rowmax16(fmaxf(s0, s1));
      float mn = fmaxf(mrun[r], mx);
      float a  = __builtin_amdgcn_exp2f((mrun[r] - mn) * LOG2E);
      float p0 = __builtin_amdgcn_exp2f((s0 - mn) * LOG2E);
      float p1 = __builtin_amdgcn_exp2f((s1 - mn) * LOG2E);
      lrun[r]  = lrun[r] * a + (p0 + p1);   // per-lane partial (2 columns)
      mrun[r]  = mn;
      av[r]    = a;
      unsigned up = pk_bf16x2(p0, p1);
      pw[(r + 8 * h) * PSTR + m]      = (unsigned short)(up & 0xFFFFu);
      pw[(r + 8 * h) * PSTR + 16 + m] = (unsigned short)(up >> 16);
    }
#pragma unroll
    for (int nc = 0; nc < 8; ++nc)
      accO[nc] = accO[nc] * av;   // vector mul -> v_pk_mul_f32 pairs

    // ---- reload P as A-fragment (same-wave DS ops are in-order) ----
    FragB pf;
    pf.q[0] = *(const u32x4*)&pw[m * PSTR + h * 8];
    pf.q[1] = *(const u32x4*)&pw[m * PSTR + 16 + h * 8];

    // ---- O += P * V ----
#pragma unroll
    for (int nc = 0; nc < 8; ++nc) {
      FragB vf;
      const unsigned short* vr = &sVt[p][(nc * 16 + m) * VSTR + h * 8];
      vf.q[0] = *(const u32x4*)vr;
      vf.q[1] = *(const u32x4*)(vr + 16);
      accO[nc] = __builtin_amdgcn_wmma_f32_16x16x32_bf16(false, pf.v, false, vf.v,
                                                         (short)0, accO[nc], false, false);
    }
    p ^= 1;
  }

  // ---- epilogue: reduce partial denominators once, then O / l ----
  float* obase = out + ((size_t)bh * SEQ + qw + 8 * h) * DHEAD + m;
#pragma unroll
  for (int r = 0; r < 8; ++r) {
    float rl = 1.f / rowsum16(lrun[r]);
#pragma unroll
    for (int nc = 0; nc < 8; ++nc)
      obase[(size_t)r * DHEAD + nc * 16] = accO[nc][r] * rl;
  }
}

extern "C" void kernel_launch(void* const* d_in, const int* in_sizes, int n_in,
                              void* d_out, int out_size, void* d_ws, size_t ws_size,
                              hipStream_t stream) {
  const float* q    = (const float*)d_in[0];
  const float* k    = (const float*)d_in[1];
  const float* v    = (const float*)d_in[2];
  const float* mask = (const float*)d_in[3];
  float* out = (float*)d_out;
  const int bh = in_sizes[0] / (SEQ * DHEAD);   // B*H = 64
  dim3 grid(SEQ / QBLK, bh);
  fa_fwd_kernel<<<grid, 256, 0, stream>>>(q, k, v, mask, out);
}
